// MultiHeadedAtention_84224308675174
// MI455X (gfx1250) — matrix operand order repaired and checked by reading, not compiled, not run on hardware
//
#include <hip/hip_runtime.h>
#include <hip/hip_bf16.h>

// ---------------------------------------------------------------------------
// MHA for MI455X (gfx1250, wave32, WMMA + TDM).  All matrix math runs on
// v_wmma_f32_16x16x32_bf16 with f32 accumulation; softmax in f32 VALU with
// wave32 shuffle reductions.  bf16 tiles are staged into LDS with
// tensor_load_to_lds (TENSORcnt) where no dtype conversion is needed; the
// TDM pad feature reproduces the bank-conflict padding of the LDS layouts.
// Shapes fixed: B=4, S=2048, DK=DV=1024, H=16, d_head=64.
// Workspace: 3x q/k/v bf16 (16 MiB each) + attn-out bf16 (16 MiB) = 64 MiB.
// ---------------------------------------------------------------------------

typedef __attribute__((ext_vector_type(16))) __bf16 bf16x16;
typedef __attribute__((ext_vector_type(8)))  float  f32x8;
typedef __attribute__((ext_vector_type(4)))  unsigned int v4u;
typedef __attribute__((ext_vector_type(8)))  int    v8i;
typedef __attribute__((ext_vector_type(4)))  int    v4i;

union AFrag { bf16x16 v; unsigned int u[8]; };

__device__ __forceinline__ unsigned short f2bf(float f) {
    unsigned int u = __float_as_uint(f);
    unsigned int r = u + 0x7FFFu + ((u >> 16) & 1u);   // round-to-nearest-even
    return (unsigned short)(r >> 16);
}

// ---------------------------------------------------------------------------
// TDM: issue a 2-D bf16 tile load (tile_d0 x tile_d1 elements) from a row-major
// tensor (stride0 elements per row) into LDS at lds_off, padding 1 DWORD after
// every 2^(pad_code+1) DWORDs (reproduces the +2-ushort padded LDS strides).
// D# bit packing per CDNA5 ISA 8.3/8.4 (group0: count/lds/global/type,
// group1: data_size/pad/dims/strides).  Issue from ONE wave only.
// 6-arg toolchain builtin: (v4u g0, v8i g1, v4i g2, v4i g3, v8i g4, i32 cpol).
// ---------------------------------------------------------------------------
__device__ __forceinline__ void tdm_load_2d(const void* gptr, unsigned lds_off,
                                            unsigned tile_d0, unsigned tile_d1,
                                            unsigned tensor_d0, unsigned tensor_d1,
                                            unsigned stride0, unsigned pad_code)
{
    unsigned long long ga = (unsigned long long)(size_t)gptr;
    v4u g0;
    g0[0] = 1u;                                        // count=1, user D#
    g0[1] = lds_off;                                   // lds_addr  (bits 63:32)
    g0[2] = (unsigned)(ga & 0xFFFFFFFFu);              // global_addr[31:0]
    g0[3] = (unsigned)((ga >> 32) & 0x01FFFFFFu)       // global_addr[56:32]
          | (2u << 30);                                // type = 2 ("image")
    v8i g1;
    g1[0] = (int)((1u << 16)                           // data_size = 2 bytes
                | (1u << 20)                           // pad_enable
                | (pad_code << 22));                   // pad_interval; pad_amount=0 (1 DW)
    g1[1] = (int)((tensor_d0 & 0xFFFFu) << 16);        // tensor_dim0[15:0]
    g1[2] = (int)(((tensor_d0 >> 16) & 0xFFFFu)        // tensor_dim0[31:16]
                | ((tensor_d1 & 0xFFFFu) << 16));      // tensor_dim1[15:0]
    g1[3] = (int)(((tensor_d1 >> 16) & 0xFFFFu)        // tensor_dim1[31:16]
                | ((tile_d0 & 0xFFFFu) << 16));        // tile_dim0
    g1[4] = (int)(tile_d1 & 0xFFFFu);                  // tile_dim1; tile_dim2=0
    g1[5] = (int)stride0;                              // tensor_dim0_stride[31:0]
    g1[6] = 0;                                         // stride0[47:32]=0, stride1=0
    g1[7] = 0;
    v4i z4 = {0, 0, 0, 0};
    v8i z8 = {0, 0, 0, 0, 0, 0, 0, 0};
    __builtin_amdgcn_tensor_load_to_lds(g0, g1, z4, z4, z8, 0);
}

// ===========================================================================
// Kernel 1: per-head input projection  y[b,h,s,e] = X[b,s,:] . W[h,:,e] + bias
// GEMM view: [8192 x 1024] x [1024 x 1024], W head-strided (H, 1024, 64).
// fp32 -> bf16 conversion happens during staging, so this stays manual.
// ===========================================================================
#define P_BM 128
#define P_BN 64
#define P_BK 32
#define P_LDA 34   // ushort stride, BK+2 pad (keeps b32 frag reads aligned)
#define P_LDB 34   // B stored transposed [n][k]

__global__ __launch_bounds__(256)
void proj_kernel(const float* __restrict__ X,     // [8192, 1024] row-major
                 const float* __restrict__ W,     // (h, d, e): h*65536 + d*64 + e
                 const float* __restrict__ bias,  // [1024]  (h*64 + e)
                 unsigned short* __restrict__ Out) // bf16 [B*H, S, 64]
{
    __shared__ unsigned short sA[P_BM * P_LDA];
    __shared__ unsigned short sB[P_BN * P_LDB];

    const int tid     = threadIdx.x;
    const int lane    = tid & 31;
    const int wave    = tid >> 5;          // 0..7
    const int col     = lane & 15;
    const int halfSel = lane >> 4;         // 0|1 (wave32 half)
    const int tileN   = blockIdx.x * P_BN;
    const int tileM   = blockIdx.y * P_BM;

    f32x8 acc[4] = {};

    for (int k0 = 0; k0 < 1024; k0 += P_BK) {
        __syncthreads();
        for (int idx = tid; idx < P_BM * P_BK; idx += 256) {
            int r  = idx >> 5;
            int kk = idx & 31;
            sA[r * P_LDA + kk] = f2bf(X[(size_t)(tileM + r) * 1024 + k0 + kk]);
        }
        for (int idx = tid; idx < P_BK * P_BN; idx += 256) {
            int kk = idx >> 6;
            int n  = idx & 63;
            int ng = tileN + n;
            float w = W[(size_t)(ng >> 6) * 65536 + (size_t)(k0 + kk) * 64 + (ng & 63)];
            sB[n * P_LDB + kk] = f2bf(w);
        }
        __syncthreads();

        AFrag a;
        {
            const unsigned short* ap = &sA[(wave * 16 + col) * P_LDA];
            const int ksel = halfSel * 16;
#pragma unroll
            for (int v = 0; v < 4; ++v) a.u[v]     = *(const unsigned int*)(ap + ksel + 2 * v);
#pragma unroll
            for (int v = 0; v < 4; ++v) a.u[4 + v] = *(const unsigned int*)(ap + ksel + 8 + 2 * v);
        }
#pragma unroll
        for (int nt = 0; nt < 4; ++nt) {
            AFrag b;
            const unsigned short* bp = &sB[(nt * 16 + col) * P_LDB + halfSel * 16];
#pragma unroll
            for (int v = 0; v < 8; ++v) b.u[v] = *(const unsigned int*)(bp + 2 * v);
            acc[nt] = __builtin_amdgcn_wmma_f32_16x16x32_bf16(
                false, a.v, false, b.v, (short)0, acc[nt], false, false);
        }
    }

    const int h = tileN >> 6;
#pragma unroll
    for (int nt = 0; nt < 4; ++nt) {
        const int   e  = nt * 16 + col;
        const float bv = bias[h * 64 + e];
#pragma unroll
        for (int g = 0; g < 8; ++g) {
            int m = tileM + wave * 16 + halfSel * 8 + g;
            int b = m >> 11, s = m & 2047;
            Out[(((size_t)b * 16 + h) * 2048 + s) * 64 + e] = f2bf(acc[nt][g] + bv);
        }
    }
}

// ===========================================================================
// Kernel 2: flash attention per (b,h).  Block = 128 thr (4 waves) x 64 q rows.
// K and V tiles (32 rows x 64 dims, bf16) are DMA'd into LDS by the Tensor
// Data Mover (wave 0 issues, TENSORcnt-synced), online softmax, P re-layout
// through per-wave LDS scratch, PV WMMAs.
// ===========================================================================
#define A_KT  32
#define A_LDK 66   // [r][d] stride, 64+2  (TDM pad: 1 DW after every 32 DW)
#define A_LDP 34   // sP [row][k], 32+2

__global__ __launch_bounds__(128)
void attn_kernel(const unsigned short* __restrict__ Q,   // bf16 [B*H, S, 64]
                 const unsigned short* __restrict__ Kb,
                 const unsigned short* __restrict__ Vb,
                 unsigned short* __restrict__ O)          // bf16 [B, S, H*64]
{
    __shared__ unsigned short sK[A_KT * A_LDK];
    __shared__ unsigned short sV[A_KT * A_LDK];
    __shared__ unsigned short sP[4 * 16 * A_LDP];         // per-wave P scratch

    const int tid     = threadIdx.x;
    const int lane    = tid & 31;
    const int wave    = tid >> 5;          // 0..3
    const int col     = lane & 15;
    const int halfSel = lane >> 4;
    const int bh      = blockIdx.y;        // 0..63
    const int b       = bh >> 4, h = bh & 15;
    const int qBase   = blockIdx.x * 64 + wave * 16;

    const unsigned short* qp = Q  + ((size_t)bh * 2048 + qBase) * 64;
    const unsigned short* kp = Kb + (size_t)bh * 2048 * 64;
    const unsigned short* vp = Vb + (size_t)bh * 2048 * 64;

    const unsigned sK_off = (unsigned)(size_t)(const void*)sK;
    const unsigned sV_off = (unsigned)(size_t)(const void*)sV;

    // q A-fragments (16 rows x 64 dims = two 16x32 frags), loaded once
    AFrag qf[2];
    {
        const unsigned short* qrow = qp + (size_t)col * 64;
        const int ksel = halfSel * 16;
#pragma unroll
        for (int c = 0; c < 2; ++c) {
#pragma unroll
            for (int v = 0; v < 4; ++v)
                qf[c].u[v]     = *(const unsigned int*)(qrow + c * 32 + ksel + 2 * v);
#pragma unroll
            for (int v = 0; v < 4; ++v)
                qf[c].u[4 + v] = *(const unsigned int*)(qrow + c * 32 + ksel + 8 + 2 * v);
        }
    }

    f32x8 o[4] = {};
    float mrow[8], lrow[8];
#pragma unroll
    for (int g = 0; g < 8; ++g) { mrow[g] = -1e30f; lrow[g] = 0.0f; }
    const float scale = 0.125f;  // 1/sqrt(64)

    for (int t = 0; t < 2048; t += A_KT) {
        __syncthreads();                    // previous tile fully consumed
        if (wave == 0) {
            tdm_load_2d(kp + (size_t)t * 64, sK_off, 64, A_KT, 64, 2048, 64, 4);
            tdm_load_2d(vp + (size_t)t * 64, sV_off, 64, A_KT, 64, 2048, 64, 4);
            __builtin_amdgcn_s_wait_tensorcnt(0);
        }
        if (t + A_KT < 2048) {              // global_prefetch_b8 on next tiles
            __builtin_prefetch(kp + (size_t)(t + A_KT) * 64 + lane * 64, 0, 1);
            __builtin_prefetch(vp + (size_t)(t + A_KT) * 64 + lane * 64, 0, 1);
        }
        __syncthreads();                    // LDS tiles visible to all waves

        // scores: 16 q-rows x 32 k-rows = two 16x16 WMMA tiles, K=64 contraction
        f32x8 s[2] = {};
#pragma unroll
        for (int nt = 0; nt < 2; ++nt) {
#pragma unroll
            for (int c = 0; c < 2; ++c) {
                AFrag kb;
                const unsigned short* bp =
                    &sK[(nt * 16 + col) * A_LDK + c * 32 + halfSel * 16];
#pragma unroll
                for (int v = 0; v < 8; ++v) kb.u[v] = *(const unsigned int*)(bp + 2 * v);
                s[nt] = __builtin_amdgcn_wmma_f32_16x16x32_bf16(
                    false, qf[c].v, false, kb.v, (short)0, s[nt], false, false);
            }
        }

        // online softmax: per-row reductions across the 16 lanes of each half
#pragma unroll
        for (int g = 0; g < 8; ++g) {
            float s0 = s[0][g] * scale;
            float s1 = s[1][g] * scale;
            float v  = fmaxf(s0, s1);
#pragma unroll
            for (int mask = 8; mask >= 1; mask >>= 1)
                v = fmaxf(v, __shfl_xor(v, mask, 32));
            float mnew  = fmaxf(mrow[g], v);
            float alpha = __expf(mrow[g] - mnew);
            float p0 = __expf(s0 - mnew);
            float p1 = __expf(s1 - mnew);
            s[0][g] = p0; s[1][g] = p1;
            float sum = p0 + p1;
#pragma unroll
            for (int mask = 8; mask >= 1; mask >>= 1)
                sum += __shfl_xor(sum, mask, 32);
            lrow[g] = lrow[g] * alpha + sum;
            mrow[g] = mnew;
#pragma unroll
            for (int e = 0; e < 4; ++e) o[e][g] *= alpha;
        }

        // C-layout P -> per-wave LDS -> A-layout frag (same-wave DS is in-order)
        unsigned short* pbase = &sP[wave * 16 * A_LDP];
#pragma unroll
        for (int nt = 0; nt < 2; ++nt)
#pragma unroll
            for (int g = 0; g < 8; ++g)
                pbase[(halfSel * 8 + g) * A_LDP + nt * 16 + col] = f2bf(s[nt][g]);

        AFrag pf;
        {
            const unsigned short* pr = pbase + col * A_LDP;
            const int ksel = halfSel * 16;
#pragma unroll
            for (int v = 0; v < 4; ++v) pf.u[v]     = *(const unsigned int*)(pr + ksel + 2 * v);
#pragma unroll
            for (int v = 0; v < 4; ++v) pf.u[4 + v] = *(const unsigned int*)(pr + ksel + 8 + 2 * v);
        }

        // O += P(16x32) x V(32x64): V B-frags are u16 gather pairs from [r][d]
#pragma unroll
        for (int e = 0; e < 4; ++e) {
            AFrag vf;
            const int nbase = e * 16 + col;
#pragma unroll
            for (int v = 0; v < 8; ++v) {
                unsigned lo = sV[(halfSel * 16 + 2 * v    ) * A_LDK + nbase];
                unsigned hi = sV[(halfSel * 16 + 2 * v + 1) * A_LDK + nbase];
                vf.u[v] = lo | (hi << 16);
            }
            o[e] = __builtin_amdgcn_wmma_f32_16x16x32_bf16(
                false, pf.v, false, vf.v, (short)0, o[e], false, false);
        }
    }

    // finalize and store bf16 to [b][s][h*64+e]
#pragma unroll
    for (int g = 0; g < 8; ++g) {
        const float inv  = 1.0f / lrow[g];
        const int   srow = qBase + halfSel * 8 + g;
#pragma unroll
        for (int e = 0; e < 4; ++e) {
            size_t oa = ((size_t)b * 2048 + srow) * 1024 + h * 64 + e * 16 + col;
            O[oa] = f2bf(o[e][g] * inv);
        }
    }
}

// ===========================================================================
// Kernel 3: output projection  out = Xa . Wo + bo   (bf16 x bf16 -> fp32)
// Xa bf16 [8192,1024] row-major: A tile staged by TDM (already bf16);
// Wo fp32 -> bf16 staging stays manual.
// ===========================================================================
__global__ __launch_bounds__(256)
void oproj_kernel(const unsigned short* __restrict__ Xa,
                  const float* __restrict__ W,
                  const float* __restrict__ bias,
                  float* __restrict__ Out)
{
    __shared__ unsigned short sA[P_BM * P_LDA];
    __shared__ unsigned short sB[P_BN * P_LDB];

    const int tid     = threadIdx.x;
    const int lane    = tid & 31;
    const int wave    = tid >> 5;
    const int col     = lane & 15;
    const int halfSel = lane >> 4;
    const int tileN   = blockIdx.x * P_BN;
    const int tileM   = blockIdx.y * P_BM;

    const unsigned sA_off = (unsigned)(size_t)(const void*)sA;

    f32x8 acc[4] = {};

    for (int k0 = 0; k0 < 1024; k0 += P_BK) {
        __syncthreads();
        if (wave == 0) {
            // 128 rows x 32 bf16; pad 1 DW after every 16 DW -> stride 34 ushorts
            tdm_load_2d(Xa + (size_t)tileM * 1024 + k0, sA_off,
                        P_BK, P_BM, 1024, 8192, 1024, 3);
        }
        for (int idx = tid; idx < P_BK * P_BN; idx += 256) {
            int kk = idx >> 6;
            int n  = idx & 63;
            sB[n * P_LDB + kk] = f2bf(W[(size_t)(k0 + kk) * 1024 + tileN + n]);
        }
        if (wave == 0) __builtin_amdgcn_s_wait_tensorcnt(0);
        __syncthreads();

        AFrag a;
        {
            const unsigned short* ap = &sA[(wave * 16 + col) * P_LDA];
        const int ksel = halfSel * 16;
#pragma unroll
            for (int v = 0; v < 4; ++v) a.u[v]     = *(const unsigned int*)(ap + ksel + 2 * v);
#pragma unroll
            for (int v = 0; v < 4; ++v) a.u[4 + v] = *(const unsigned int*)(ap + ksel + 8 + 2 * v);
        }
#pragma unroll
        for (int nt = 0; nt < 4; ++nt) {
            AFrag b;
            const unsigned short* bp = &sB[(nt * 16 + col) * P_LDB + halfSel * 16];
#pragma unroll
            for (int v = 0; v < 8; ++v) b.u[v] = *(const unsigned int*)(bp + 2 * v);
            acc[nt] = __builtin_amdgcn_wmma_f32_16x16x32_bf16(
                false, a.v, false, b.v, (short)0, acc[nt], false, false);
        }
    }

#pragma unroll
    for (int nt = 0; nt < 4; ++nt) {
        const int   n  = tileN + nt * 16 + col;
        const float bv = bias[n];
#pragma unroll
        for (int g = 0; g < 8; ++g) {
            int m = tileM + wave * 16 + halfSel * 8 + g;
            Out[(size_t)m * 1024 + n] = acc[nt][g] + bv;
        }
    }
}

// ===========================================================================
extern "C" void kernel_launch(void* const* d_in, const int* in_sizes, int n_in,
                              void* d_out, int out_size, void* d_ws, size_t ws_size,
                              hipStream_t stream)
{
    (void)in_sizes; (void)n_in; (void)out_size; (void)ws_size;

    const float* Query = (const float*)d_in[0];
    const float* Key   = (const float*)d_in[1];
    const float* Value = (const float*)d_in[2];
    const float* Wq    = (const float*)d_in[3];
    const float* bq    = (const float*)d_in[4];
    const float* Wk    = (const float*)d_in[5];
    const float* bk    = (const float*)d_in[6];
    const float* Wv    = (const float*)d_in[7];
    const float* bv    = (const float*)d_in[8];
    const float* Wo    = (const float*)d_in[9];
    const float* bo    = (const float*)d_in[10];

    const size_t NTOK = (size_t)8192 * 1024;      // 8.4M bf16 elements each
    unsigned short* q_ws = (unsigned short*)d_ws;
    unsigned short* k_ws = q_ws + NTOK;
    unsigned short* v_ws = k_ws + NTOK;
    unsigned short* a_ws = v_ws + NTOK;           // total 64 MiB of workspace

    dim3 gProj(16, 64);                            // N-tiles x M-tiles
    proj_kernel<<<gProj, 256, 0, stream>>>(Query, Wq, bq, q_ws);
    proj_kernel<<<gProj, 256, 0, stream>>>(Key,   Wk, bk, k_ws);
    proj_kernel<<<gProj, 256, 0, stream>>>(Value, Wv, bv, v_ws);

    dim3 gAttn(32, 64);                            // q-row tiles x (B*H)
    attn_kernel<<<gAttn, 128, 0, stream>>>(q_ws, k_ws, v_ws, a_ws);

    oproj_kernel<<<gProj, 256, 0, stream>>>(a_ws, Wo, bo, (float*)d_out);
}
